// WPHOp_57080115363929
// MI455X (gfx1250) — matrix operand order repaired and checked by reading, not compiled, not run on hardware
//
#include <hip/hip_runtime.h>
#include <hip/hip_bf16.h>
#include <math.h>

// ---------------- problem constants (match reference) ----------------
#define BATCH   8
#define MDIM    256
#define NDIM    256
#define MN      (MDIM * NDIM)       // 65536
#define PPSI    24
#define FPHI    4
#define QPSI    96                  // 24 * 4 harmonics
#define QPHI_PAD 16                 // 4 * 2 harmonics = 8, padded to 16 for WMMA tile
#define KWPH    2000
#define KSM     12
#define NCOEF   (KWPH + KSM)        // 2012

// Gram kernel chunking: K processed in chunks of KCH via TDM double buffering
#define KCH     1024
#define NCH     (MN / KCH)          // 64 (even: required by smem reuse below)
#define STRIPB  (16 * KCH * 2)      // bytes per 16-row bf16 strip = 32 KB
#define GRAM_LDS (2 * 4 * STRIPB)   // 2 buffers x 4 strips = 256 KB

typedef __attribute__((ext_vector_type(16))) __bf16 v16bf;
typedef __attribute__((ext_vector_type(8)))  float  v8f;
typedef __attribute__((ext_vector_type(4)))  unsigned v4u;
typedef __attribute__((ext_vector_type(8)))  unsigned v8u;

__device__ __forceinline__ unsigned brev8(unsigned x) { return __brev(x) >> 24; }

// ---------------- 256-point in-LDS radix-2 FFT -----------------------
__device__ void fft256_stages(float* re, float* im, int t, float sign) {
  for (int s = 0; s < 8; ++s) {
    int half = 1 << s;
    int grp  = t >> s;
    int pos  = t & (half - 1);
    int i0   = (grp << (s + 1)) + pos;
    int i1   = i0 + half;
    float ang = sign * 6.28318530717958647692f * (float)pos / (float)(half << 1);
    float sn, cs;
    __sincosf(ang, &sn, &cs);
    float xr = re[i1], xi = im[i1];
    float tr = xr * cs - xi * sn;
    float ti = xr * sn + xi * cs;
    float ur = re[i0], ui = im[i0];
    re[i0] = ur + tr; im[i0] = ui + ti;
    re[i1] = ur - tr; im[i1] = ui - ti;
    __syncthreads();
  }
}

// ---------------- K1: forward FFT over rows, real input --------------
__global__ void k_fft_rows_r2c(const float* __restrict__ x, float2* __restrict__ xf) {
  __shared__ float sre[2][256];
  __shared__ float sim[2][256];
  int line = threadIdx.x >> 7;
  int t    = threadIdx.x & 127;
  int row  = blockIdx.x * 2 + line;
  int b    = blockIdx.z;
  const float* src = x + ((size_t)b * MDIM + row) * NDIM;
  float* re = sre[line]; float* im = sim[line];
  int e0 = t, e1 = t + 128;
  re[brev8(e0)] = src[e0]; im[brev8(e0)] = 0.f;
  re[brev8(e1)] = src[e1]; im[brev8(e1)] = 0.f;
  __syncthreads();
  fft256_stages(re, im, t, -1.f);
  float2* dst = xf + ((size_t)b * MDIM + row) * NDIM;
  dst[e0] = make_float2(re[e0], im[e0]);
  dst[e1] = make_float2(re[e1], im[e1]);
}

// ---------------- K2: FFT over columns, in-place complex -------------
__global__ void k_fft_cols(float2* __restrict__ data, float sign, float scale) {
  __shared__ float sre[2][256];
  __shared__ float sim[2][256];
  int line = threadIdx.x >> 7;
  int t    = threadIdx.x & 127;
  int col  = blockIdx.x * 2 + line;
  int b    = blockIdx.z;
  float2* base = data + (size_t)b * MN + col;
  int e0 = t, e1 = t + 128;
  float2 v0 = base[(size_t)e0 * NDIM];
  float2 v1 = base[(size_t)e1 * NDIM];
  float* re = sre[line]; float* im = sim[line];
  re[brev8(e0)] = v0.x; im[brev8(e0)] = v0.y;
  re[brev8(e1)] = v1.x; im[brev8(e1)] = v1.y;
  __syncthreads();
  fft256_stages(re, im, t, sign);
  base[(size_t)e0 * NDIM] = make_float2(re[e0] * scale, im[e0] * scale);
  base[(size_t)e1 * NDIM] = make_float2(re[e1] * scale, im[e1] * scale);
}

// ---------------- K3: filter multiply + inverse FFT over rows --------
__global__ void k_filt_ifft_rows(const float2* __restrict__ xf, const float* __restrict__ filt,
                                 float2* __restrict__ tmp, int nfilt) {
  __shared__ float sre[2][256];
  __shared__ float sim[2][256];
  int line = threadIdx.x >> 7;
  int t    = threadIdx.x & 127;
  int row  = blockIdx.x * 2 + line;
  int p    = blockIdx.y;
  int b    = blockIdx.z;
  const float2* src = xf   + ((size_t)b * MDIM + row) * NDIM;
  const float*  fl  = filt + ((size_t)p * MDIM + row) * NDIM;
  float* re = sre[line]; float* im = sim[line];
  int e0 = t, e1 = t + 128;
  float2 a0 = src[e0]; float f0 = fl[e0];
  float2 a1 = src[e1]; float f1 = fl[e1];
  re[brev8(e0)] = a0.x * f0; im[brev8(e0)] = a0.y * f0;
  re[brev8(e1)] = a1.x * f1; im[brev8(e1)] = a1.y * f1;
  __syncthreads();
  fft256_stages(re, im, t, 1.f);
  const float scale = 1.f / 256.f;
  float2* dst = tmp + (((size_t)b * nfilt + p) * MDIM + row) * NDIM;
  dst[e0] = make_float2(re[e0] * scale, im[e0] * scale);
  dst[e1] = make_float2(re[e1] * scale, im[e1] * scale);
}

// ---------------- K4: inverse FFT over cols + phase harmonics --------
// Writes bf16 real/imag planes H[b, q, kk] with kk = col*256 + m.
__global__ void k_ifft_cols_harm(const float2* __restrict__ tmp,
                                 __hip_bfloat16* __restrict__ Hr,
                                 __hip_bfloat16* __restrict__ Hi,
                                 int nfilt, int nh,
                                 float h0, float h1, float h2, float h3,
                                 int Qpad) {
  __shared__ float sre[2][256];
  __shared__ float sim[2][256];
  int line = threadIdx.x >> 7;
  int t    = threadIdx.x & 127;
  int col  = blockIdx.x * 2 + line;
  int p    = blockIdx.y;
  int b    = blockIdx.z;
  const float2* base = tmp + ((size_t)b * nfilt + p) * MN + col;
  float* re = sre[line]; float* im = sim[line];
  int e0 = t, e1 = t + 128;
  float2 v0 = base[(size_t)e0 * NDIM];
  float2 v1 = base[(size_t)e1 * NDIM];
  re[brev8(e0)] = v0.x; im[brev8(e0)] = v0.y;
  re[brev8(e1)] = v1.x; im[brev8(e1)] = v1.y;
  __syncthreads();
  fft256_stages(re, im, t, 1.f);
  const float scale = 1.f / 256.f;
  float harms[4] = {h0, h1, h2, h3};
  for (int ei = 0; ei < 2; ++ei) {
    int m = t + ei * 128;
    float zr = re[m] * scale, zi = im[m] * scale;
    float r  = sqrtf(zr * zr + zi * zi);
    float th = atan2f(zi, zr);
    size_t kk = (size_t)col * NDIM + m;
    for (int h = 0; h < nh; ++h) {
      float sn, cs;
      __sincosf(harms[h] * th, &sn, &cs);
      int q = p * nh + h;
      size_t off = ((size_t)b * Qpad + q) * MN + kk;
      Hr[off] = __float2bfloat16(r * cs);
      Hi[off] = __float2bfloat16(r * sn);
    }
  }
}

// ---------------- K5: per-plane mean (fp32 accumulation) -------------
__global__ void k_mean(const __hip_bfloat16* __restrict__ Hr,
                       const __hip_bfloat16* __restrict__ Hi,
                       float* __restrict__ mu, int Qpad) {
  int q = blockIdx.x, b = blockIdx.y;
  size_t base = ((size_t)b * Qpad + q) * MN;
  float sr = 0.f, si = 0.f;
  for (int i = threadIdx.x; i < MN; i += 256) {
    sr += __bfloat162float(Hr[base + i]);
    si += __bfloat162float(Hi[base + i]);
  }
  __shared__ float rr[256], ri[256];
  rr[threadIdx.x] = sr; ri[threadIdx.x] = si;
  __syncthreads();
  for (int s = 128; s > 0; s >>= 1) {
    if (threadIdx.x < (unsigned)s) {
      rr[threadIdx.x] += rr[threadIdx.x + s];
      ri[threadIdx.x] += ri[threadIdx.x + s];
    }
    __syncthreads();
  }
  if (threadIdx.x == 0) {
    const float inv = 1.f / (float)MN;
    mu[((size_t)b * Qpad + q) * 2 + 0] = rr[0] * inv;
    mu[((size_t)b * Qpad + q) * 2 + 1] = ri[0] * inv;
  }
}

// ---------------- K6: zero helper (pad rows of phi planes) -----------
__global__ void k_zero_u32(unsigned* __restrict__ p, size_t n) {
  size_t i = (size_t)blockIdx.x * blockDim.x + threadIdx.x;
  size_t stride = (size_t)gridDim.x * blockDim.x;
  for (; i < n; i += stride) p[i] = 0u;
}

// ---------------- TDM: DMA one 16-row x KCH-col bf16 strip to LDS ----
// 2D tensor descriptor (D# groups 0 and 1, ISA 8.3/8.4):
//   data_size=2B, tile = KCH x 16, row stride = MN elements.
// Issued once per wave (EXEC ignored); tracked by TENSORcnt.
__device__ __forceinline__ void tdm_load_strip(const __hip_bfloat16* gsrc, unsigned lds_off) {
  unsigned long long ga = (unsigned long long)(size_t)gsrc;
  v4u g0;
  g0.x = 1u;                                              // count=1 (valid user D#)
  g0.y = lds_off;                                         // lds_addr (bytes)
  g0.z = (unsigned)(ga & 0xFFFFFFFFu);                    // global_addr[31:0]
  g0.w = (unsigned)((ga >> 32) & 0x01FFFFFFu) | (2u << 30); // addr[56:32] | type=2
  v8u g1;
  g1.s0 = (1u << 16);                                     // wg_mask=0, data_size=1 (2B)
  g1.s1 = ((unsigned)KCH & 0xFFFFu) << 16;                // tensor_dim0[15:0] @ 63:48
  g1.s2 = ((unsigned)KCH >> 16) | (16u << 16);            // tensor_dim0 hi | tensor_dim1 lo
  g1.s3 = ((unsigned)KCH << 16);                          // tensor_dim1 hi | tile_dim0
  g1.s4 = 16u;                                            // tile_dim1=16, tile_dim2=0
  g1.s5 = (unsigned)MN;                                   // tensor_dim0_stride[31:0]
  g1.s6 = 0u;                                             // stride hi | dim1_stride lo
  g1.s7 = 0u;                                             // dim1_stride hi
  asm volatile("tensor_load_to_lds %0, %1" :: "s"(g0), "s"(g1) : "memory");
}

// ---------------- K7: complex Gram via bf16 WMMA + TDM staging -------
// G = H * H^H per batch. Block = (qt, rt, b); 8 waves split K inside a chunk.
// Double-buffered: wave0 DMAs next chunk's 4 strips (Ar/Ai/Br/Bi) while all
// waves run WMMAs on the current chunk out of LDS.
//   Gr = Hr*Hr^T + Hi*Hi^T ; Gi = Hi*Hr^T - Hr*Hi^T
__global__ void k_gram_wmma(const __hip_bfloat16* __restrict__ Hr,
                            const __hip_bfloat16* __restrict__ Hi,
                            float* __restrict__ Gr, float* __restrict__ Gi, int Q) {
  extern __shared__ char smem[];   // GRAM_LDS bytes: [buf][strip(Ar,Ai,Br,Bi)]
  int qt = blockIdx.x, rt = blockIdx.y, b = blockIdx.z;
  int tid = threadIdx.x;
  int w = tid >> 5;                // wave 0..7
  int l = tid & 31;                // lane
  int hiHalf = l >> 4;             // 0 or 1

  const __hip_bfloat16* rowsA_r = Hr + ((size_t)b * Q + qt * 16) * MN;
  const __hip_bfloat16* rowsA_i = Hi + ((size_t)b * Q + qt * 16) * MN;
  const __hip_bfloat16* rowsB_r = Hr + ((size_t)b * Q + rt * 16) * MN;
  const __hip_bfloat16* rowsB_i = Hi + ((size_t)b * Q + rt * 16) * MN;

  v8f accRR = {}; v8f accII = {}; v8f accIR = {}; v8f accRI = {};
  union U { uint4 q[2]; v16bf v; };

  // prologue: DMA chunk 0 into buffer 0
  if (w == 0) {
    tdm_load_strip(rowsA_r, 0 * STRIPB);
    tdm_load_strip(rowsA_i, 1 * STRIPB);
    tdm_load_strip(rowsB_r, 2 * STRIPB);
    tdm_load_strip(rowsB_i, 3 * STRIPB);
  }

  for (int n = 0; n < NCH; ++n) {
    if (w == 0) __builtin_amdgcn_s_wait_tensorcnt(0);  // chunk n landed
    __syncthreads();                                   // publish buf[n&1]; prev compute done
    if (w == 0 && n + 1 < NCH) {                       // prefetch chunk n+1
      unsigned bo = (unsigned)(((n + 1) & 1) * 4 * STRIPB);
      size_t kc = (size_t)(n + 1) * KCH;
      tdm_load_strip(rowsA_r + kc, bo + 0 * STRIPB);
      tdm_load_strip(rowsA_i + kc, bo + 1 * STRIPB);
      tdm_load_strip(rowsB_r + kc, bo + 2 * STRIPB);
      tdm_load_strip(rowsB_i + kc, bo + 3 * STRIPB);
    }
    // compute on buf[n&1]: per wave, K sub-stripe of KCH/8 = 128 -> 4 WMMA steps
    char* buf = smem + (size_t)(n & 1) * 4 * STRIPB;
    const uint4* sAr = (const uint4*)(buf + 0 * STRIPB);
    const uint4* sAi = (const uint4*)(buf + 1 * STRIPB);
    const uint4* sBr = (const uint4*)(buf + 2 * STRIPB);
    const uint4* sBi = (const uint4*)(buf + 3 * STRIPB);
    int rowOff = (l & 15) * KCH;   // element offset of this lane's row in strip
    #pragma unroll
    for (int ks = w * (KCH / 8); ks < (w + 1) * (KCH / 8); ks += 32) {
      // A fragment: K = ks + hiHalf*8 + {0..7}, ks + 16 + hiHalf*8 + {0..7}
      int ia = (rowOff + ks + hiHalf * 8) >> 3;
      // B fragment: K = ks + hiHalf*8 + {0..7}, ks + 16 + hiHalf*8 + {0..7}
      int ib = (rowOff + ks + hiHalf * 8) >> 3;
      U ar, ai, br, bi;
      ar.q[0] = sAr[ia]; ar.q[1] = sAr[ia + 2];
      ai.q[0] = sAi[ia]; ai.q[1] = sAi[ia + 2];
      br.q[0] = sBr[ib]; br.q[1] = sBr[ib + 2];
      bi.q[0] = sBi[ib]; bi.q[1] = sBi[ib + 2];
      accRR = __builtin_amdgcn_wmma_f32_16x16x32_bf16(false, ar.v, false, br.v,
                                                      (short)0, accRR, false, false);
      accII = __builtin_amdgcn_wmma_f32_16x16x32_bf16(false, ai.v, false, bi.v,
                                                      (short)0, accII, false, false);
      accIR = __builtin_amdgcn_wmma_f32_16x16x32_bf16(false, ai.v, false, br.v,
                                                      (short)0, accIR, false, false);
      accRI = __builtin_amdgcn_wmma_f32_16x16x32_bf16(false, ar.v, false, bi.v,
                                                      (short)0, accRI, false, false);
    }
  }

  // cross-wave reduction of the 16x16 f32 tile; reuse buf0 region of LDS
  // (last chunk index NCH-1 is odd -> final compute read buf1, buf0 is idle).
  float* redR = (float*)smem;
  float* redI = redR + 8 * 256;
  #pragma unroll
  for (int e = 0; e < 8; ++e) {
    redR[w * 256 + l * 8 + e] = accRR[e] + accII[e];
    redI[w * 256 + l * 8 + e] = accIR[e] - accRI[e];
  }
  __syncthreads();
  float sr = 0.f, si = 0.f;
  #pragma unroll
  for (int w2 = 0; w2 < 8; ++w2) {
    sr += redR[w2 * 256 + tid];
    si += redI[w2 * 256 + tid];
  }
  // C/D layout: lane (l2), VGPR (e2): M = e2 + 8*(l2>=16), N = l2 & 15
  int l2 = tid >> 3, e2 = tid & 7;
  int Mr = e2 + ((l2 >= 16) ? 8 : 0);
  int Nc = l2 & 15;
  size_t o = ((size_t)b * Q + (qt * 16 + Mr)) * Q + (rt * 16 + Nc);
  Gr[o] = sr;
  Gi[o] = si;
}

// ---------------- K8: gather coefficients ----------------------------
__global__ void k_gather(const float* __restrict__ GpsiR, const float* __restrict__ GpsiI,
                         const float* __restrict__ muPsi,
                         const float* __restrict__ GphiR, const float* __restrict__ GphiI,
                         const float* __restrict__ muPhi,
                         const int* __restrict__ wi1, const int* __restrict__ wi2,
                         const int* __restrict__ si1, const int* __restrict__ si2,
                         float* __restrict__ out) {
  int j = blockIdx.x * blockDim.x + threadIdx.x;
  if (j >= BATCH * NCOEF) return;
  int b = j / NCOEF, k = j % NCOEF;
  const float invMN = 1.f / (float)MN;
  float gr, gi, m1r, m1i, m2r, m2i;
  if (k < KWPH) {
    int i1 = wi1[k], i2 = wi2[k];
    gr = GpsiR[((size_t)b * QPSI + i1) * QPSI + i2] * invMN;
    gi = GpsiI[((size_t)b * QPSI + i1) * QPSI + i2] * invMN;
    m1r = muPsi[((size_t)b * QPSI + i1) * 2 + 0];
    m1i = muPsi[((size_t)b * QPSI + i1) * 2 + 1];
    m2r = muPsi[((size_t)b * QPSI + i2) * 2 + 0];
    m2i = muPsi[((size_t)b * QPSI + i2) * 2 + 1];
  } else {
    int kk = k - KWPH;
    int i1 = si1[kk], i2 = si2[kk];
    gr = GphiR[((size_t)b * QPHI_PAD + i1) * QPHI_PAD + i2] * invMN;
    gi = GphiI[((size_t)b * QPHI_PAD + i1) * QPHI_PAD + i2] * invMN;
    m1r = muPhi[((size_t)b * QPHI_PAD + i1) * 2 + 0];
    m1i = muPhi[((size_t)b * QPHI_PAD + i1) * 2 + 1];
    m2r = muPhi[((size_t)b * QPHI_PAD + i2) * 2 + 0];
    m2i = muPhi[((size_t)b * QPHI_PAD + i2) * 2 + 1];
  }
  float cr = m1r * m2r + m1i * m2i;   // mu1 * conj(mu2)
  float ci = m1i * m2r - m1r * m2i;
  out[(size_t)j * 2 + 0] = gr - cr;
  out[(size_t)j * 2 + 1] = gi - ci;
}

// ---------------- host-side orchestration ----------------------------
extern "C" void kernel_launch(void* const* d_in, const int* in_sizes, int n_in,
                              void* d_out, int out_size, void* d_ws, size_t ws_size,
                              hipStream_t stream) {
  (void)in_sizes; (void)n_in; (void)out_size; (void)ws_size;
  const float* x     = (const float*)d_in[0];
  const float* psi_f = (const float*)d_in[1];
  const float* phi_f = (const float*)d_in[2];
  const int*   wi1   = (const int*)d_in[3];
  const int*   wi2   = (const int*)d_in[4];
  const int*   si1   = (const int*)d_in[5];
  const int*   si2   = (const int*)d_in[6];
  float* out = (float*)d_out;

  // workspace carve-out (~340 MB total)
  char* ws = (char*)d_ws;
  size_t off = 0;
  auto alloc = [&](size_t bytes) -> void* {
    void* p = ws + off;
    off += (bytes + 255) & ~(size_t)255;
    return p;
  };
  float2* xf   = (float2*)alloc(sizeof(float2) * (size_t)BATCH * MN);
  float2* tmp  = (float2*)alloc(sizeof(float2) * (size_t)BATCH * PPSI * MN);  // reused for phi
  __hip_bfloat16* HpsiR = (__hip_bfloat16*)alloc(2ull * BATCH * QPSI * MN);
  __hip_bfloat16* HpsiI = (__hip_bfloat16*)alloc(2ull * BATCH * QPSI * MN);
  __hip_bfloat16* HphiR = (__hip_bfloat16*)alloc(2ull * BATCH * QPHI_PAD * MN);
  __hip_bfloat16* HphiI = (__hip_bfloat16*)alloc(2ull * BATCH * QPHI_PAD * MN);
  float* muPsi = (float*)alloc(sizeof(float) * (size_t)BATCH * QPSI * 2);
  float* muPhi = (float*)alloc(sizeof(float) * (size_t)BATCH * QPHI_PAD * 2);
  float* GpsiR = (float*)alloc(sizeof(float) * (size_t)BATCH * QPSI * QPSI);
  float* GpsiI = (float*)alloc(sizeof(float) * (size_t)BATCH * QPSI * QPSI);
  float* GphiR = (float*)alloc(sizeof(float) * (size_t)BATCH * QPHI_PAD * QPHI_PAD);
  float* GphiI = (float*)alloc(sizeof(float) * (size_t)BATCH * QPHI_PAD * QPHI_PAD);

  // x_f = fft2(x)
  k_fft_rows_r2c<<<dim3(128, 1, BATCH), 256, 0, stream>>>(x, xf);
  k_fft_cols<<<dim3(128, 1, BATCH), 256, 0, stream>>>(xf, -1.f, 1.f);

  // ---- psi branch ----
  k_filt_ifft_rows<<<dim3(128, PPSI, BATCH), 256, 0, stream>>>(xf, psi_f, tmp, PPSI);
  k_ifft_cols_harm<<<dim3(128, PPSI, BATCH), 256, 0, stream>>>(
      tmp, HpsiR, HpsiI, PPSI, 4, 0.f, 1.f, 2.f, 4.f, QPSI);
  k_mean<<<dim3(QPSI, BATCH), 256, 0, stream>>>(HpsiR, HpsiI, muPsi, QPSI);
  k_gram_wmma<<<dim3(QPSI / 16, QPSI / 16, BATCH), 256, GRAM_LDS, stream>>>(
      HpsiR, HpsiI, GpsiR, GpsiI, QPSI);

  // ---- phi branch (Q padded 8 -> 16; zero the planes first) ----
  size_t nphiU32 = (size_t)BATCH * QPHI_PAD * MN / 2;
  k_zero_u32<<<4096, 256, 0, stream>>>((unsigned*)HphiR, nphiU32);
  k_zero_u32<<<4096, 256, 0, stream>>>((unsigned*)HphiI, nphiU32);
  k_filt_ifft_rows<<<dim3(128, FPHI, BATCH), 256, 0, stream>>>(xf, phi_f, tmp, FPHI);
  k_ifft_cols_harm<<<dim3(128, FPHI, BATCH), 256, 0, stream>>>(
      tmp, HphiR, HphiI, FPHI, 2, 0.f, 1.f, 0.f, 0.f, QPHI_PAD);
  k_mean<<<dim3(QPHI_PAD, BATCH), 256, 0, stream>>>(HphiR, HphiI, muPhi, QPHI_PAD);
  k_gram_wmma<<<dim3(1, 1, BATCH), 256, GRAM_LDS, stream>>>(
      HphiR, HphiI, GphiR, GphiI, QPHI_PAD);

  // ---- gather coefficients ----
  k_gather<<<(BATCH * NCOEF + 255) / 256, 256, 0, stream>>>(
      GpsiR, GpsiI, muPsi, GphiR, GphiI, muPhi, wi1, wi2, si1, si2, out);
}
